// MultiHeadQKVAttention_70531952935642
// MI455X (gfx1250) — matrix operand order repaired and checked by reading, not compiled, not run on hardware
//
#include <hip/hip_runtime.h>

// Problem constants (reference: B=2, N=M=2048, H=8, d_head=32, model=256)
#define BB 2
#define NN 2048
#define MM 2048
#define HH 8
#define DDH 32
#define DK 256

typedef __attribute__((ext_vector_type(16))) __bf16 v16bf;
typedef __attribute__((ext_vector_type(8)))  float  v8f;

union V16 {
    v16bf v;
    uint4 q[2];
    unsigned u[8];
};

// fp32 -> bf16 via hardware converter (v_cvt_pk_bf16_f32 on gfx1250)
__device__ __forceinline__ unsigned short f2bfu(float f) {
    __bf16 h = (__bf16)f;
    return __builtin_bit_cast(unsigned short, h);
}

__device__ __forceinline__ unsigned pack2(float a, float b) {
    return (unsigned)f2bfu(a) | ((unsigned)f2bfu(b) << 16);
}

// ---------------------------------------------------------------------------
// Weight transpose + bf16 convert: W[k][n] (fp32) -> Wt[n][k] (bf16)
// ---------------------------------------------------------------------------
__global__ void convt_kernel(const float* __restrict__ W,
                             unsigned short* __restrict__ Wt) {
    int idx = blockIdx.x * blockDim.x + threadIdx.x;   // 65536 elements
    int n = idx >> 8;
    int k = idx & 255;
    Wt[idx] = f2bfu(W[k * DK + n]);
}

// ---------------------------------------------------------------------------
// Elementwise fp32 -> bf16 (packed pairs), same row-major layout.
// ---------------------------------------------------------------------------
__global__ void cvt_x_kernel(const float* __restrict__ X,
                             unsigned short* __restrict__ Xb) {
    int idx = blockIdx.x * blockDim.x + threadIdx.x;   // 524288 pairs
    float2 f = ((const float2*)X)[idx];
    ((unsigned*)Xb)[idx] = pack2(f.x, f.y);
}

// ---------------------------------------------------------------------------
// Unified 16x16-tile GEMM: Xbf[4096][256] (bf16) @ Wt (bf16, n-major) + bias
//   mode 0: bf16 dst[((b*H+h)*Seq + s)*32 + d]   (Q, K layout; d contiguous)
//   mode 1: bf16 dst[((b*H+h)*32 + d)*M + s]     (V transposed; m contiguous)
//   mode 2: fp32 out[row][256]                   (final projection)
// One wave computes a 16x16 output tile; K-loop of 8 x (K=32) WMMAs.
// ---------------------------------------------------------------------------
__global__ __launch_bounds__(128) void gemm_kernel(
    const unsigned short* __restrict__ Xbf,
    const unsigned short* __restrict__ Wt,
    const float* __restrict__ bias,
    unsigned short* __restrict__ dst_bf,
    float* __restrict__ dst_f32,
    int mode) {
    const int wave = threadIdx.x >> 5;
    const int lane = threadIdx.x & 31;
    const int li = lane & 15;
    const int hi = lane >> 4;
    const int tile = blockIdx.x * 4 + wave;   // 4096 tiles
    const int colT = tile & 15;
    const int rowT = tile >> 4;
    const int arow = rowT * 16 + li;          // A-operand row for this lane
    const int ncol = colT * 16 + li;          // B-operand column for this lane

    v8f acc = {};
    const unsigned short* xr = Xbf + arow * DK;
    const unsigned short* wr = Wt + ncol * DK;
    #pragma unroll
    for (int k0 = 0; k0 < DK; k0 += 32) {
        V16 a, b;
        const unsigned short* p0 = xr + k0 + hi * 8;   // K groups hi*8, 16+hi*8
        a.q[0] = *(const uint4*)(p0);
        a.q[1] = *(const uint4*)(p0 + 16);
        const unsigned short* wp = wr + k0 + hi * 8;   // K groups hi*8, 16+hi*8
        b.q[0] = *(const uint4*)(wp);
        b.q[1] = *(const uint4*)(wp + 16);
        acc = __builtin_amdgcn_wmma_f32_16x16x32_bf16(
            false, a.v, false, b.v, (short)0, acc, false, false);
    }

    const float bv = bias[ncol];
    const int h  = ncol >> 5;
    const int d0 = ncol & 31;
    #pragma unroll
    for (int r = 0; r < 8; ++r) {
        int xrow = rowT * 16 + r + hi * 8;    // D-layout: M = r + 8*hi
        float val = acc[r] + bv;
        if (mode == 2) {
            dst_f32[(size_t)xrow * DK + ncol] = val;
        } else {
            int bi = xrow / NN;
            int si = xrow % NN;
            unsigned short hv = f2bfu(val);
            if (mode == 0)
                dst_bf[(((bi * HH + h) * NN) + si) * DDH + d0] = hv;
            else
                dst_bf[(((bi * HH + h) * DDH) + d0) * MM + si] = hv;
        }
    }
}

// ---------------------------------------------------------------------------
// Flash attention. One wave owns a 16-query tile of one (b,h).
//   S^T = K_tile(16m x 32d) @ Q^T(32d x 16n)  -> lane=n, VGPR r = m_local
//   online softmax over m (per-lane reduce + shfl_xor(16) with partner half)
//   P^T staged in LDS as [n][32m] bf16  ->  B-operand for
//   O^T += V^T(16d x 32m) @ P^T(32m x 16n)    (two d-halves)
// ---------------------------------------------------------------------------
__global__ __launch_bounds__(128) void attn_kernel(
    const unsigned short* __restrict__ q_bf,   // [B*H][N][32]
    const unsigned short* __restrict__ k_bf,   // [B*H][M][32]
    const unsigned short* __restrict__ vt_bf,  // [B*H][32][M]
    const float* __restrict__ presence,        // [B][M]
    unsigned short* __restrict__ o_bf) {       // [B*N][256]
    __shared__ unsigned short lds_p[4][16 * 32];

    const int wave = threadIdx.x >> 5;
    const int lane = threadIdx.x & 31;
    const int li = lane & 15;
    const int hi = lane >> 4;
    const int tile = blockIdx.x * 4 + wave;    // 2048 tiles
    const int nT = tile & (NN / 16 - 1);
    const int bh = tile >> 7;
    const int b  = bh >> 3;
    const int h  = bh & 7;

    // Q B-operand, loaded once: lane = column n, d groups at hi*8 and 16+hi*8
    const unsigned short* qrow = q_bf + (bh * NN + nT * 16 + li) * DDH;
    V16 qB;
    qB.q[0] = *(const uint4*)(qrow + hi * 8);
    qB.q[1] = *(const uint4*)(qrow + 16 + hi * 8);

    const unsigned short* kbase = k_bf + (size_t)bh * MM * DDH;
    const unsigned short* vtb   = vt_bf + (size_t)bh * DDH * MM;
    const float* pres = presence + b * MM;

    float run_max = -__builtin_inff();
    float row_l = 0.0f;
    v8f o1 = {}, o2 = {};
    const float scale = 0.17677669529663687f;  // 1/sqrt(32)
    unsigned short* myP = lds_p[wave];

    for (int m0 = 0; m0 < MM; m0 += 32) {
        if (m0 + 32 < MM) {   // gfx1250 global_prefetch for next chunk
            __builtin_prefetch(kbase + (m0 + 32 + li) * DDH, 0, 1);
            __builtin_prefetch(vtb + (size_t)li * MM + m0 + 32, 0, 1);
        }
        // K A-tiles (rows = m, K = d): groups at d = hi*8 and 16+hi*8
        V16 kA1, kA2;
        const unsigned short* kr1 = kbase + (m0 + li) * DDH;
        const unsigned short* kr2 = kbase + (m0 + 16 + li) * DDH;
        kA1.q[0] = *(const uint4*)(kr1 + hi * 8);
        kA1.q[1] = *(const uint4*)(kr1 + 16 + hi * 8);
        kA2.q[0] = *(const uint4*)(kr2 + hi * 8);
        kA2.q[1] = *(const uint4*)(kr2 + 16 + hi * 8);

        v8f z = {};
        v8f c1 = __builtin_amdgcn_wmma_f32_16x16x32_bf16(
            false, kA1.v, false, qB.v, (short)0, z, false, false);
        v8f c2 = __builtin_amdgcn_wmma_f32_16x16x32_bf16(
            false, kA2.v, false, qB.v, (short)0, z, false, false);

        // mask + scale, tile max
        float s1[8], s2[8];
        float tmax = -__builtin_inff();
        #pragma unroll
        for (int r = 0; r < 8; ++r) {
            int m1 = m0 + hi * 8 + r;          // C-layout: M = r + 8*hi
            int m2 = m1 + 16;
            float mk1 = (1.0f - pres[m1]) * 1e32f;
            float mk2 = (1.0f - pres[m2]) * 1e32f;
            s1[r] = (c1[r] - mk1) * scale;
            s2[r] = (c2[r] - mk2) * scale;
            tmax = fmaxf(tmax, fmaxf(s1[r], s2[r]));
        }
        tmax = fmaxf(tmax, __shfl_xor(tmax, 16, 32));
        float nmax = fmaxf(run_max, tmax);
        float corr = __expf(run_max - nmax);
        run_max = nmax;

        // exp, row-sum, stage P^T into LDS as [n][32m] bf16
        float lsum = 0.0f;
        unsigned short* prow = myP + li * 32;
        #pragma unroll
        for (int r = 0; r < 8; r += 2) {
            float pa = __expf(s1[r] - nmax), pb = __expf(s1[r + 1] - nmax);
            float pc = __expf(s2[r] - nmax), pd = __expf(s2[r + 1] - nmax);
            lsum += pa + pb + pc + pd;
            *(unsigned*)(prow + hi * 8 + r)      = pack2(pa, pb);
            *(unsigned*)(prow + 16 + hi * 8 + r) = pack2(pc, pd);
        }
        lsum += __shfl_xor(lsum, 16, 32);
        row_l = row_l * corr + lsum;
        #pragma unroll
        for (int r = 0; r < 8; ++r) { o1[r] *= corr; o2[r] *= corr; }

        // wave-internal LDS ordering (cross-lane exchange through DS unit)
        asm volatile("s_wait_dscnt 0" ::: "memory");

        // P^T as B-operand: lane = n, m groups at hi*8 and 16+hi*8
        V16 pB;
        pB.q[0] = *(const uint4*)(myP + li * 32 + hi * 8);
        pB.q[1] = *(const uint4*)(myP + li * 32 + 16 + hi * 8);

        // V^T A-tiles (rows = d, K = m): groups at m = m0+hi*8, m0+16+hi*8
        V16 vA1, vA2;
        const unsigned short* v1 = vtb + (size_t)li * MM + m0;
        const unsigned short* v2 = vtb + (size_t)(16 + li) * MM + m0;
        vA1.q[0] = *(const uint4*)(v1 + hi * 8);
        vA1.q[1] = *(const uint4*)(v1 + 16 + hi * 8);
        vA2.q[0] = *(const uint4*)(v2 + hi * 8);
        vA2.q[1] = *(const uint4*)(v2 + 16 + hi * 8);

        o1 = __builtin_amdgcn_wmma_f32_16x16x32_bf16(
            false, vA1.v, false, pB.v, (short)0, o1, false, false);
        o2 = __builtin_amdgcn_wmma_f32_16x16x32_bf16(
            false, vA2.v, false, pB.v, (short)0, o2, false, false);
    }

    // normalize and store O (lane = n fixed; d = r + 8*hi [+16 for tile 2])
    float inv_l = 1.0f / row_l;
    const int n = nT * 16 + li;
    unsigned short* orow = o_bf + (size_t)(b * NN + n) * DK + h * DDH;
    #pragma unroll
    for (int r = 0; r < 8; r += 2) {
        *(unsigned*)(orow + hi * 8 + r) =
            pack2(o1[r] * inv_l, o1[r + 1] * inv_l);
        *(unsigned*)(orow + 16 + hi * 8 + r) =
            pack2(o2[r] * inv_l, o2[r + 1] * inv_l);
    }
}

// ---------------------------------------------------------------------------
// Host launcher
// ---------------------------------------------------------------------------
extern "C" void kernel_launch(void* const* d_in, const int* in_sizes, int n_in,
                              void* d_out, int out_size, void* d_ws,
                              size_t ws_size, hipStream_t stream) {
    const float* queries  = (const float*)d_in[0];
    const float* keys     = (const float*)d_in[1];
    const float* values   = (const float*)d_in[2];
    const float* presence = (const float*)d_in[3];
    const float* Wq = (const float*)d_in[4];
    const float* bq = (const float*)d_in[5];
    const float* Wk = (const float*)d_in[6];
    const float* bk = (const float*)d_in[7];
    const float* Wv = (const float*)d_in[8];
    const float* bv = (const float*)d_in[9];
    const float* Wo = (const float*)d_in[10];
    const float* bo = (const float*)d_in[11];
    float* out = (float*)d_out;

    char* ws = (char*)d_ws;
    const size_t WSZ = 256 * 256 * 2;                    // 128 KB per weight
    const size_t TSZ = (size_t)BB * HH * NN * DDH * 2;   // 2 MB per tensor
    unsigned short* wq_t = (unsigned short*)(ws);
    unsigned short* wk_t = (unsigned short*)(ws + WSZ);
    unsigned short* wv_t = (unsigned short*)(ws + 2 * WSZ);
    unsigned short* wo_t = (unsigned short*)(ws + 3 * WSZ);
    unsigned short* q_bf  = (unsigned short*)(ws + 4 * WSZ);
    unsigned short* k_bf  = (unsigned short*)(ws + 4 * WSZ + TSZ);
    unsigned short* vt_bf = (unsigned short*)(ws + 4 * WSZ + 2 * TSZ);
    unsigned short* o_bf  = (unsigned short*)(ws + 4 * WSZ + 3 * TSZ);
    unsigned short* x_bf  = (unsigned short*)(ws + 4 * WSZ + 4 * TSZ);

    convt_kernel<<<256, 256, 0, stream>>>(Wq, wq_t);
    convt_kernel<<<256, 256, 0, stream>>>(Wk, wk_t);
    convt_kernel<<<256, 256, 0, stream>>>(Wv, wv_t);
    convt_kernel<<<256, 256, 0, stream>>>(Wo, wo_t);

    // Q projection (x_bf scratch reused; stream order serializes)
    cvt_x_kernel<<<2048, 256, 0, stream>>>(queries, x_bf);
    gemm_kernel<<<1024, 128, 0, stream>>>(x_bf, wq_t, bq, q_bf, nullptr, 0);
    // K projection
    cvt_x_kernel<<<2048, 256, 0, stream>>>(keys, x_bf);
    gemm_kernel<<<1024, 128, 0, stream>>>(x_bf, wk_t, bk, k_bf, nullptr, 0);
    // V projection (transposed per-head layout)
    cvt_x_kernel<<<2048, 256, 0, stream>>>(values, x_bf);
    gemm_kernel<<<1024, 128, 0, stream>>>(x_bf, wv_t, bv, vt_bf, nullptr, 1);

    attn_kernel<<<512, 128, 0, stream>>>(q_bf, k_bf, vt_bf, presence, o_bf);

    // Output projection -> fp32
    gemm_kernel<<<1024, 128, 0, stream>>>(o_bf, wo_t, bo, nullptr, out, 2);
}
